// SelfAttention_62483184222663
// MI455X (gfx1250) — compile-verified, matrix operand order checked by reading, not run on hardware
//
#include <hip/hip_runtime.h>
#include <hip/hip_bf16.h>
#include <stdint.h>

typedef __attribute__((ext_vector_type(16))) __bf16 v16bf;
typedef __attribute__((ext_vector_type(8)))  __bf16 v8bf;
typedef __attribute__((ext_vector_type(8)))  float  v8f;
typedef __attribute__((ext_vector_type(4)))  unsigned int u32x4;
typedef __attribute__((ext_vector_type(8)))  int          i32x8;
typedef __attribute__((ext_vector_type(4)))  int          i32x4;

#define EMB 2048
#define NHEADS 16
#define NKV 4
#define HEADD 128
#define GRP 4
#define BATCH 2
#define SEQ 2048
#define MTOT (BATCH*SEQ)   /* 4096 rows of x */
#define KVDIM (NKV*HEADD)  /* 512 */

__device__ __forceinline__ __bf16 f2bf(float f) {
  uint32_t u = __builtin_bit_cast(uint32_t, f);
  uint32_t r = u + 0x7FFFu + ((u >> 16) & 1u);   // round-to-nearest-even
  return __builtin_bit_cast(__bf16, (uint16_t)(r >> 16));
}
__device__ __forceinline__ float bf2f(__bf16 h) {
  uint16_t s = __builtin_bit_cast(uint16_t, h);
  return __builtin_bit_cast(float, ((uint32_t)s) << 16);
}

// Low 32 bits of a flat shared-memory pointer are the LDS byte address
// (ISA: flat->LDS aperture mapping truncates to addr[31:0]).
__device__ __forceinline__ uint32_t lds_addr_of(const void* p) {
  return (uint32_t)(uintptr_t)p;
}

// ------------------------------------------------------------------ TDM issue
// 2D tile load global -> LDS via Tensor Data Mover.
// tile: tile1 rows x tile0 elements (bf16, 2 bytes), row stride = stride0 elems.
__device__ __forceinline__ void tdm_load_2d(uint32_t lds_addr, const void* gptr,
                                            uint32_t tdim0, uint32_t tdim1,
                                            uint32_t tile0, uint32_t tile1,
                                            uint64_t stride0) {
  uint64_t ga = (uint64_t)(uintptr_t)gptr;
  u32x4 g0;
  g0[0] = 1u;                                   // count=1 (valid user D#)
  g0[1] = lds_addr;                             // lds_addr [63:32]
  g0[2] = (uint32_t)(ga & 0xFFFFFFFFu);         // global_addr [95:64]
  g0[3] = (uint32_t)((ga >> 32) & 0x01FFFFFFu)  // global_addr [120:96]
        | (2u << 30);                           // type=2 ("image")
  i32x8 g1;
  g1[0] = (int)(1u << 16);                      // data_size=1 (2 bytes), mask=0
  g1[1] = (int)((tdim0 & 0xFFFFu) << 16);       // tensor_dim0[15:0] @bits48
  g1[2] = (int)((tdim0 >> 16) | ((tdim1 & 0xFFFFu) << 16));
  g1[3] = (int)((tdim1 >> 16) | (tile0 << 16)); // tile_dim0 @bits112
  g1[4] = (int)(tile1 & 0xFFFFu);               // tile_dim1; tile_dim2=0
  g1[5] = (int)(uint32_t)(stride0 & 0xFFFFFFFFu);      // dim0 stride lo
  g1[6] = (int)(uint32_t)((stride0 >> 32) & 0xFFFFu);  // stride hi; stride1=0
  g1[7] = 0;
  i32x4 z4 = {0, 0, 0, 0};                      // 2D: groups 2/3 unused
  i32x8 z8 = {0, 0, 0, 0, 0, 0, 0, 0};
  __builtin_amdgcn_tensor_load_to_lds(g0, g1, z4, z4, z8, 0);
}

// ---------------------------------------------------------------- f32 -> bf16
__global__ void cvt_f32_bf16(const float* __restrict__ src,
                             __bf16* __restrict__ dst, int n) {
  int i = blockIdx.x * blockDim.x + threadIdx.x;
  if (i < n) dst[i] = f2bf(src[i]);
}

// ---------------------------------------------------------------- NT GEMM
// C[M,N] = A[M,K] * B[N,K]^T, K-major both sides (einsum('me,ne->mn')).
// Block tile 128x64, 8 waves (4m x 2n), each wave 32x32 = 4 WMMA tiles.
// TDM double-buffered staging of A (128x32) and B (64x32) tiles.
#define BK 32

__global__ __launch_bounds__(256) void gemm_nt_bf16(
    const __bf16* __restrict__ A, const __bf16* __restrict__ B,
    void* __restrict__ Cv, int M, int N, int K, int c_f32)
{
  __shared__ __align__(128) __bf16 As[2][128 * BK];
  __shared__ __align__(128) __bf16 Bs[2][64 * BK];
  const int t    = threadIdx.x;
  const int lane = t & 31;
  const int w    = t >> 5;
  const int wm   = (w & 3) * 32;     // 4 wave rows of 32
  const int wn   = (w >> 2) * 32;    // 2 wave cols of 32
  const int m0   = blockIdx.y * 128;
  const int n0   = blockIdx.x * 64;
  const int ln   = lane & 15;
  const int hi   = lane >> 4;
  const int kb   = hi * 8;           // K sub-base per lane half (ISA 7.12.2)

  v8f zero = {};
  v8f acc[2][2];
  acc[0][0] = zero; acc[0][1] = zero; acc[1][0] = zero; acc[1][1] = zero;

  const int nk = K / BK;
  auto issue = [&](int ik, int buf) {
    if (w == 0) {
      int kk = ik * BK;
      tdm_load_2d(lds_addr_of(&As[buf][0]), A + (size_t)m0 * K + kk,
                  (uint32_t)K, (uint32_t)M, BK, 128, (uint64_t)K);
      tdm_load_2d(lds_addr_of(&Bs[buf][0]), B + (size_t)n0 * K + kk,
                  (uint32_t)K, (uint32_t)N, BK, 64, (uint64_t)K);
    }
  };

  issue(0, 0);
  for (int ik = 0; ik < nk; ++ik) {
    const int buf = ik & 1;
    if (ik + 1 < nk) {
      issue(ik + 1, buf ^ 1);
      if (w == 0) __builtin_amdgcn_s_wait_tensorcnt(2);  // current pair resident
    } else {
      if (w == 0) __builtin_amdgcn_s_wait_tensorcnt(0);
    }
    __syncthreads();

    const __bf16* as = &As[buf][0];
    const __bf16* bs = &Bs[buf][0];
    v16bf af[2], bf[2];
    #pragma unroll
    for (int i = 0; i < 8; ++i) {
      af[0][i]     = as[(wm + ln) * BK + kb + i];
      af[0][i + 8] = as[(wm + ln) * BK + kb + 16 + i];
      af[1][i]     = as[(wm + 16 + ln) * BK + kb + i];
      af[1][i + 8] = as[(wm + 16 + ln) * BK + kb + 16 + i];
      bf[0][i]     = bs[(wn + ln) * BK + kb + i];
      bf[0][i + 8] = bs[(wn + ln) * BK + kb + 16 + i];
      bf[1][i]     = bs[(wn + 16 + ln) * BK + kb + i];
      bf[1][i + 8] = bs[(wn + 16 + ln) * BK + kb + 16 + i];
    }
    #pragma unroll
    for (int i = 0; i < 2; ++i)
      #pragma unroll
      for (int j = 0; j < 2; ++j)
        acc[i][j] = __builtin_amdgcn_wmma_f32_16x16x32_bf16(
            false, af[i], false, bf[j], (short)0, acc[i][j], false, false);
    __syncthreads();   // all waves done with buf before TDM overwrites it
  }

  #pragma unroll
  for (int i = 0; i < 2; ++i) {
    #pragma unroll
    for (int j = 0; j < 2; ++j) {
      #pragma unroll
      for (int r = 0; r < 8; ++r) {
        int mrow = m0 + wm + i * 16 + r + 8 * hi;
        int ncol = n0 + wn + j * 16 + ln;
        if (c_f32) ((float*)Cv)[(size_t)mrow * N + ncol] = acc[i][j][r];
        else       ((__bf16*)Cv)[(size_t)mrow * N + ncol] = f2bf(acc[i][j][r]);
      }
    }
  }
}

// ---------------------------------------------------------------- RoPE (in place, bf16)
__global__ void rope_kernel(__bf16* __restrict__ P, int heads, int n) {
  int idx = blockIdx.x * blockDim.x + threadIdx.x;
  if (idx >= n) return;
  int i    = idx & 63;
  int rest = idx >> 6;
  int head = rest % heads;
  int bs   = rest / heads;
  int s    = bs & (SEQ - 1);
  float fr  = __expf(-9.210340372f * (float)(2 * i) * (1.0f / 128.0f));
  float ang = (float)s * 6.283185307f * fr;
  float sn, c;
  __sincosf(ang, &sn, &c);
  size_t base = ((size_t)bs * heads + head) * HEADD + 2 * i;
  float a = bf2f(P[base]), b = bf2f(P[base + 1]);
  P[base]     = f2bf(a * c - b * sn);
  P[base + 1] = f2bf(a * sn + b * c);
}

// ---------------------------------------------------------------- flash attention
// One wave per (batch, head, 16-row q tile). KV chunks of 32, TDM double-buffered.
__global__ __launch_bounds__(32) void flash_attn(
    const __bf16* __restrict__ Q, const __bf16* __restrict__ Kg,
    const __bf16* __restrict__ Vg, __bf16* __restrict__ Z,
    const int* __restrict__ causalPtr)
{
  __shared__ __align__(128) __bf16 Ks[2][32 * HEADD];
  __shared__ __align__(128) __bf16 Vs[2][32 * HEADD];
  __shared__ __align__(16)  __bf16 Ps[16 * 32];
  __shared__ float Ssc[16 * 32];
  __shared__ float alphaS[16];
  __shared__ float linvS[16];

  const int lane = threadIdx.x & 31;
  const int ln = lane & 15, hi = lane >> 4, kb = hi * 8;
  const int q0  = blockIdx.x * 16;
  const int h   = blockIdx.y;
  const int b   = blockIdx.z;
  const int kvh = h / GRP;
  const int causal = *causalPtr;
  const float scale = 0.08838834764831845f;   // 1/sqrt(128)

  // Q fragments: 16x128 bf16 as 4 A-operands of K=32
  v16bf qf[4];
  {
    const __bf16* qrow = Q + ((size_t)(b * SEQ) + q0 + ln) * EMB + h * HEADD;
    #pragma unroll
    for (int c = 0; c < 4; ++c) {
      v8bf lo  = *(const v8bf*)(qrow + c * 32 + kb);
      v8bf hi8 = *(const v8bf*)(qrow + c * 32 + kb + 16);
      #pragma unroll
      for (int i = 0; i < 8; ++i) { qf[c][i] = lo[i]; qf[c][i + 8] = hi8[i]; }
    }
  }

  v8f zero = {};
  v8f acc[8];
  #pragma unroll
  for (int t = 0; t < 8; ++t) acc[t] = zero;
  float rm = -1e30f, rl = 0.0f;

  const int nch = causal ? (q0 + 16 + 31) / 32 : SEQ / 32;
  auto issueKV = [&](int ch, int buf) {
    int kv0 = ch * 32;
    const __bf16* kt = Kg + ((size_t)(b * SEQ) + kv0) * KVDIM + kvh * HEADD;
    const __bf16* vt = Vg + ((size_t)(b * SEQ) + kv0) * KVDIM + kvh * HEADD;
    tdm_load_2d(lds_addr_of(&Ks[buf][0]), kt, KVDIM, SEQ, HEADD, 32, KVDIM);
    tdm_load_2d(lds_addr_of(&Vs[buf][0]), vt, KVDIM, SEQ, HEADD, 32, KVDIM);
  };

  issueKV(0, 0);
  for (int ch = 0; ch < nch; ++ch) {
    const int buf = ch & 1;
    const int kv0 = ch * 32;
    if (ch + 1 < nch) {
      issueKV(ch + 1, buf ^ 1);
      __builtin_amdgcn_s_wait_tensorcnt(2);   // current K/V pair resident
    } else {
      __builtin_amdgcn_s_wait_tensorcnt(0);
    }
    __syncthreads();

    // S = Q K^T : two 16x16 tiles, contraction over D=128 (4 WMMAs each)
    #pragma unroll
    for (int n16 = 0; n16 < 2; ++n16) {
      v8f sacc = zero;
      #pragma unroll
      for (int c = 0; c < 4; ++c) {
        v16bf kf;
        const __bf16* krow = &Ks[buf][(n16 * 16 + ln) * HEADD + c * 32];
        #pragma unroll
        for (int i = 0; i < 8; ++i) {
          kf[i]     = krow[kb + i];
          kf[i + 8] = krow[kb + 16 + i];
        }
        sacc = __builtin_amdgcn_wmma_f32_16x16x32_bf16(false, qf[c], false, kf,
                                                       (short)0, sacc, false, false);
      }
      #pragma unroll
      for (int r = 0; r < 8; ++r)
        Ssc[(r + 8 * hi) * 32 + n16 * 16 + ln] = sacc[r] * scale;
    }
    __syncthreads();

    // online softmax: lanes 0..15 own one row each
    if (lane < 16) {
      const int m = lane, sq = q0 + m;
      float vbuf[32];
      float mx = rm;
      #pragma unroll
      for (int j = 0; j < 32; ++j) {
        int skv = kv0 + j;
        float v = (!causal || skv <= sq) ? Ssc[m * 32 + j] : -1e30f;
        vbuf[j] = v;
        mx = fmaxf(mx, v);
      }
      float al  = __expf(rm - mx);
      float sum = rl * al;
      #pragma unroll
      for (int j = 0; j < 32; ++j) {
        float p = __expf(vbuf[j] - mx);
        Ps[m * 32 + j] = f2bf(p);
        sum += p;
      }
      rm = mx; rl = sum; alphaS[m] = al;
    }
    __syncthreads();

    // O = alpha*O + P V : 8 d-tiles of 16, contraction over kv=32
    v16bf pf;
    #pragma unroll
    for (int i = 0; i < 8; ++i) {
      pf[i]     = Ps[ln * 32 + kb + i];
      pf[i + 8] = Ps[ln * 32 + kb + 16 + i];
    }
    #pragma unroll
    for (int t = 0; t < 8; ++t) {
      #pragma unroll
      for (int r = 0; r < 8; ++r) acc[t][r] *= alphaS[r + 8 * hi];
      v16bf vf;
      #pragma unroll
      for (int i = 0; i < 8; ++i) {
        vf[i]     = Vs[buf][(kb + i) * HEADD + t * 16 + ln];
        vf[i + 8] = Vs[buf][(kb + 16 + i) * HEADD + t * 16 + ln];
      }
      acc[t] = __builtin_amdgcn_wmma_f32_16x16x32_bf16(false, pf, false, vf,
                                                       (short)0, acc[t], false, false);
    }
    __syncthreads();
  }

  if (lane < 16) linvS[lane] = 1.0f / rl;
  __syncthreads();
  #pragma unroll
  for (int t = 0; t < 8; ++t) {
    #pragma unroll
    for (int r = 0; r < 8; ++r) {
      int m = r + 8 * hi;
      float o = acc[t][r] * linvS[m];
      Z[((size_t)(b * SEQ) + q0 + m) * EMB + h * HEADD + t * 16 + ln] = f2bf(o);
    }
  }
}

// ---------------------------------------------------------------- launch
extern "C" void kernel_launch(void* const* d_in, const int* in_sizes, int n_in,
                              void* d_out, int out_size, void* d_ws, size_t ws_size,
                              hipStream_t stream) {
  const float* x  = (const float*)d_in[0];
  const float* Wq = (const float*)d_in[1];
  const float* Wk = (const float*)d_in[2];
  const float* Wv = (const float*)d_in[3];
  const float* Wo = (const float*)d_in[4];
  const int* causal = (const int*)d_in[5];

  char* ws = (char*)d_ws;
  size_t off = 0;
  auto alloc = [&](size_t elems) {
    __bf16* p = (__bf16*)(ws + off);
    off += ((elems * 2 + 255) & ~(size_t)255);
    return p;
  };
  __bf16* xb  = alloc((size_t)MTOT * EMB);
  __bf16* wqb = alloc((size_t)EMB * EMB);
  __bf16* wkb = alloc((size_t)KVDIM * EMB);
  __bf16* wvb = alloc((size_t)KVDIM * EMB);
  __bf16* wob = alloc((size_t)EMB * EMB);
  __bf16* Qb  = alloc((size_t)MTOT * EMB);
  __bf16* Kb  = alloc((size_t)MTOT * KVDIM);
  __bf16* Vb  = alloc((size_t)MTOT * KVDIM);
  __bf16* Zb  = alloc((size_t)MTOT * EMB);

  auto cvt = [&](const float* s, __bf16* d, int n) {
    cvt_f32_bf16<<<(n + 255) / 256, 256, 0, stream>>>(s, d, n);
  };
  cvt(x,  xb,  MTOT * EMB);
  cvt(Wq, wqb, EMB * EMB);
  cvt(Wk, wkb, KVDIM * EMB);
  cvt(Wv, wvb, KVDIM * EMB);
  cvt(Wo, wob, EMB * EMB);

  // Q/K/V projections (bf16 out)
  gemm_nt_bf16<<<dim3(EMB / 64,   MTOT / 128), 256, 0, stream>>>(xb, wqb, Qb, MTOT, EMB,   EMB, 0);
  gemm_nt_bf16<<<dim3(KVDIM / 64, MTOT / 128), 256, 0, stream>>>(xb, wkb, Kb, MTOT, KVDIM, EMB, 0);
  gemm_nt_bf16<<<dim3(KVDIM / 64, MTOT / 128), 256, 0, stream>>>(xb, wvb, Vb, MTOT, KVDIM, EMB, 0);

  // RoPE on Q and K
  rope_kernel<<<(MTOT * NHEADS * 64 + 255) / 256, 256, 0, stream>>>(Qb, NHEADS, MTOT * NHEADS * 64);
  rope_kernel<<<(MTOT * NKV    * 64 + 255) / 256, 256, 0, stream>>>(Kb, NKV,    MTOT * NKV    * 64);

  // attention
  flash_attn<<<dim3(SEQ / 16, NHEADS, BATCH), 32, 0, stream>>>(Qb, Kb, Vb, Zb, causal);

  // output projection (f32 out, straight to d_out)
  gemm_nt_bf16<<<dim3(EMB / 64, MTOT / 128), 256, 0, stream>>>(Zb, wob, d_out, MTOT, EMB, EMB, 1);
}
